// Attention_12386685682126
// MI455X (gfx1250) — matrix operand order repaired and checked by reading, not compiled, not run on hardware
//
#include <hip/hip_runtime.h>
#include <hip/hip_bf16.h>

// ---------------------------------------------------------------------------
// Types for CDNA5 WMMA (wave32). v16bf = A/B fragment, v8f = f32 C/D fragment.
// ---------------------------------------------------------------------------
typedef __bf16 bf16_t;
typedef __attribute__((ext_vector_type(16))) __bf16 v16bf;
typedef __attribute__((ext_vector_type(8)))  __bf16 v8bf;
typedef __attribute__((ext_vector_type(8)))  float  v8f;

#define B_BATCH 4
#define SEQ     2048
#define DMODEL  1024
#define NHEADS  16
#define HDIM    64
#define LDQKV   3072   // 3 * NHEADS * HDIM
#define NEG_BIG (-1.0e30f)
// 1/sqrt(HDIM) * log2(e): folded into Q so WMMA scores land in exp2 domain.
#define Q_PRESCALE (0.125f * 1.44269504f)

static __device__ __forceinline__ v8f wmma_bf16_f32(v16bf a, v16bf b, v8f c) {
    // D = A(16x32 bf16) * B(32x16 bf16) + C(16x16 f32)
    return __builtin_amdgcn_wmma_f32_16x16x32_bf16(
        /*neg_a=*/false, a, /*neg_b=*/false, b,
        /*c_mod=*/(short)0, c, /*reuse_a=*/false, /*reuse_b=*/false);
}

// ---------------------------------------------------------------------------
// Cross-lane xor within 16-lane groups, VALU-only (DPP8 / DPP16 row_ror:8),
// avoiding ds_bpermute round-trips through the LDS pipe.
// ---------------------------------------------------------------------------
constexpr int dpp8sel(int a, int b, int c, int d, int e, int f, int g, int h) {
    return a | (b << 3) | (c << 6) | (d << 9) | (e << 12) | (f << 15) |
           (g << 18) | (h << 21);
}

template <int M>
static __device__ __forceinline__ float lane_xor(float x) {
#if __has_builtin(__builtin_amdgcn_mov_dpp8) && __has_builtin(__builtin_amdgcn_update_dpp)
    const int i = __float_as_int(x);
    int r;
    if constexpr (M == 1)
        r = __builtin_amdgcn_mov_dpp8(i, dpp8sel(1, 0, 3, 2, 5, 4, 7, 6));
    else if constexpr (M == 2)
        r = __builtin_amdgcn_mov_dpp8(i, dpp8sel(2, 3, 0, 1, 6, 7, 4, 5));
    else if constexpr (M == 4)
        r = __builtin_amdgcn_mov_dpp8(i, dpp8sel(4, 5, 6, 7, 0, 1, 2, 3));
    else  // M == 8 : rotate by 8 within a 16-lane row == lane ^ 8
        r = __builtin_amdgcn_update_dpp(0, i, 0x128 /*ROW_ROR:8*/, 0xF, 0xF, true);
    return __int_as_float(r);
#else
    return __shfl_xor(x, M, 32);
#endif
}

static __device__ __forceinline__ float red_max16(float v) {
    v = fmaxf(v, lane_xor<1>(v));
    v = fmaxf(v, lane_xor<2>(v));
    v = fmaxf(v, lane_xor<4>(v));
    v = fmaxf(v, lane_xor<8>(v));
    return v;
}
static __device__ __forceinline__ float red_sum16(float v) {
    v += lane_xor<1>(v);
    v += lane_xor<2>(v);
    v += lane_xor<4>(v);
    v += lane_xor<8>(v);
    return v;
}

// A fragment: row-major X[M,K]; lane holds row (lane&15), k chunks at
// (lane>>4)*8 and 16+(lane>>4)*8 (per ISA 16-bit A 16x32 layout).
static __device__ __forceinline__ v16bf load_frag_a(const bf16_t* X, int ld,
                                                    int m0, int k0) {
    const int lane = threadIdx.x & 31;
    const int hi   = lane >> 4;
    const bf16_t* p = X + (size_t)(m0 + (lane & 15)) * ld + k0 + hi * 8;
    v8bf lo = *(const v8bf*)(p);
    v8bf hh = *(const v8bf*)(p + 16);
    return __builtin_shufflevector(lo, hh, 0,1,2,3,4,5,6,7,8,9,10,11,12,13,14,15);
}

// B fragment for y = X * W^T with W[N,K] row-major: B[k][n] = W[n][k].
// Lane holds column n=(lane&15), k chunks at (lane>>4)*8 and 16+(lane>>4)*8.
static __device__ __forceinline__ v16bf load_frag_b(const bf16_t* W, int ld,
                                                    int n0, int k0) {
    const int lane = threadIdx.x & 31;
    const bf16_t* p = W + (size_t)(n0 + (lane & 15)) * ld + k0 + (lane >> 4) * 8;
    v8bf lo = *(const v8bf*)(p);
    v8bf hh = *(const v8bf*)(p + 16);
    return __builtin_shufflevector(lo, hh, 0,1,2,3,4,5,6,7,8,9,10,11,12,13,14,15);
}

// ---------------------------------------------------------------------------
// Kernel 1: f32 -> bf16 cast (weights)
// ---------------------------------------------------------------------------
__global__ void cast_f32_bf16_kernel(const float* __restrict__ src,
                                     bf16_t* __restrict__ dst, int n) {
    int i = blockIdx.x * blockDim.x + threadIdx.x;
    const int stride = gridDim.x * blockDim.x;
    for (; i < n; i += stride) dst[i] = (bf16_t)src[i];
}

// ---------------------------------------------------------------------------
// Kernel 2: LayerNorm over last dim (1024) + bf16 cast. One block per row.
// ---------------------------------------------------------------------------
__global__ __launch_bounds__(256)
void layernorm_bf16_kernel(const float* __restrict__ x,
                           const float* __restrict__ gamma,
                           const float* __restrict__ beta,
                           bf16_t* __restrict__ xn) {
    const int row = blockIdx.x;          // 8192 rows
    const int t   = threadIdx.x;         // 256 threads * 4 = 1024
    const float4 v = ((const float4*)(x + (size_t)row * DMODEL))[t];
    __shared__ float rs[256], rss[256];
    rs[t]  = v.x + v.y + v.z + v.w;
    rss[t] = v.x*v.x + v.y*v.y + v.z*v.z + v.w*v.w;
    __syncthreads();
#pragma unroll
    for (int off = 128; off > 0; off >>= 1) {
        if (t < off) { rs[t] += rs[t + off]; rss[t] += rss[t + off]; }
        __syncthreads();
    }
    const float mu   = rs[0] * (1.0f / DMODEL);
    const float var  = rss[0] * (1.0f / DMODEL) - mu * mu;
    const float rstd = rsqrtf(var + 1e-5f);
    const float4 gv = ((const float4*)gamma)[t];
    const float4 bv = ((const float4*)beta)[t];
    bf16_t* o = xn + (size_t)row * DMODEL + t * 4;
    o[0] = (bf16_t)((v.x - mu) * rstd * gv.x + bv.x);
    o[1] = (bf16_t)((v.y - mu) * rstd * gv.y + bv.y);
    o[2] = (bf16_t)((v.z - mu) * rstd * gv.z + bv.z);
    o[3] = (bf16_t)((v.w - mu) * rstd * gv.w + bv.w);
}

// ---------------------------------------------------------------------------
// Kernel 3: transpose the V part of qkv into Vt[b,h, hd, seq] (bf16) so the
// P@V B-fragments become contiguous 32B loads. LDS-tiled 32x32 transpose.
// ---------------------------------------------------------------------------
__global__ __launch_bounds__(256)
void transpose_v_kernel(const bf16_t* __restrict__ qkv, bf16_t* __restrict__ vt) {
    __shared__ bf16_t tile[32][33];
    const int bh = blockIdx.z;             // bb*16 + hh
    const int bb = bh >> 4, hh = bh & 15;
    const int s0 = blockIdx.x * 32;        // seq tile
    const int h0 = blockIdx.y * 32;        // hd tile
    const int tx = threadIdx.x, ty = threadIdx.y;   // 32 x 8
    const bf16_t* src = qkv + (size_t)bb * SEQ * LDQKV + 2 * DMODEL + hh * HDIM;
#pragma unroll
    for (int j = ty; j < 32; j += 8)
        tile[j][tx] = src[(size_t)(s0 + j) * LDQKV + h0 + tx];
    __syncthreads();
    bf16_t* dst = vt + (size_t)bh * HDIM * SEQ;
#pragma unroll
    for (int j = ty; j < 32; j += 8)
        dst[(size_t)(h0 + j) * SEQ + s0 + tx] = tile[tx][j];
}

// ---------------------------------------------------------------------------
// Kernel 4/6: WMMA GEMM  C[M,N] = A[M,K] * B[N,K]^T  (torch-Linear shape).
// Block = 256 threads = 8 waves; each wave owns a 32(M) x 64(N) strip
// (8 f32 accumulators, B-fragments shared across both M halves);
// block tile = 256M x 64N. Columns < scale_cols get multiplied by col_scale
// in the epilogue (used to fold softmax scale * log2(e) into Q).
// ---------------------------------------------------------------------------
template <bool OUT_BF16>
__global__ __launch_bounds__(256)
void wmma_gemm_nt_kernel(const bf16_t* __restrict__ A,
                         const bf16_t* __restrict__ B,
                         void* __restrict__ Cout,
                         int M, int N, int K,
                         int scale_cols, float col_scale) {
    const int wave = threadIdx.x >> 5;
    const int m0   = blockIdx.y * 256 + wave * 32;
    const int n0   = blockIdx.x * 64;
    const int lane = threadIdx.x & 31;
    const int hi   = lane >> 4;
    const int col  = lane & 15;

    v8f acc[2][4] = {};
    for (int k0 = 0; k0 < K; k0 += 32) {
        if (k0 + 32 < K)  // global_prefetch of next A chunk
            __builtin_prefetch(A + (size_t)(m0 + col) * K + k0 + 32, 0, 1);
        const v16bf a0 = load_frag_a(A, K, m0, k0);
        const v16bf a1 = load_frag_a(A, K, m0 + 16, k0);
#pragma unroll
        for (int t = 0; t < 4; ++t) {
            const v16bf bfrag = load_frag_b(B, K, n0 + t * 16, k0);
            acc[0][t] = wmma_bf16_f32(a0, bfrag, acc[0][t]);
            acc[1][t] = wmma_bf16_f32(a1, bfrag, acc[1][t]);
        }
    }
#pragma unroll
    for (int t = 0; t < 4; ++t) {
        const float sc = (n0 + t * 16) < scale_cols ? col_scale : 1.0f;
#pragma unroll
        for (int h = 0; h < 2; ++h) {
#pragma unroll
            for (int e = 0; e < 8; ++e) {
                const size_t idx =
                    (size_t)(m0 + h * 16 + hi * 8 + e) * N + (n0 + t * 16 + col);
                if (OUT_BF16) ((bf16_t*)Cout)[idx] = (bf16_t)(acc[h][t][e] * sc);
                else          ((float*)Cout)[idx]  = acc[h][t][e] * sc;
            }
        }
    }
}

// ---------------------------------------------------------------------------
// Kernel 5: flash attention. One wave handles one (batch, head, 32-query)
// tile; K/V fragments are shared across both 16-row halves. Q was pre-scaled
// by scale*log2(e) in the QKV GEMM, so scores are already in the exp2 domain.
//   qkv: [b, n, 3, H, 64] bf16 (row stride 3072);  vt: [b,h, hd, seq] bf16
//   out: [b, n, H*64] bf16
// ---------------------------------------------------------------------------
__global__ __launch_bounds__(256)
void flash_attn_kernel(const bf16_t* __restrict__ qkv,
                       const bf16_t* __restrict__ vt,
                       bf16_t* __restrict__ attn_out) {
    const int wave = threadIdx.x >> 5;
    const int lane = threadIdx.x & 31;
    const int hi   = lane >> 4;
    const int col  = lane & 15;

    const int task = blockIdx.x * 8 + wave;   // 4096 tasks total
    const int qt   = task & 63;               // 64 q-tiles of 32
    const int hh   = (task >> 6) & 15;        // head
    const int bb   = task >> 10;              // batch
    const int q0   = qt * 32;

    const bf16_t* base = qkv + (size_t)bb * SEQ * LDQKV;
    const bf16_t* Qb = base + hh * HDIM;
    const bf16_t* Kb = base + DMODEL + hh * HDIM;
    const bf16_t* Vt = vt + (size_t)(bb * NHEADS + hh) * HDIM * SEQ;

    __shared__ bf16_t lds_p[8][2][16 * 32];   // per-wave P-tile bounce (16KB)

    v16bf aq[2][2];
#pragma unroll
    for (int h = 0; h < 2; ++h) {
        aq[h][0] = load_frag_a(Qb, LDQKV, q0 + 16 * h, 0);
        aq[h][1] = load_frag_a(Qb, LDQKV, q0 + 16 * h, 32);
    }

    v8f o[2][4] = {};
    float rm[2][8], rl[2][8];
#pragma unroll
    for (int h = 0; h < 2; ++h)
#pragma unroll
        for (int e = 0; e < 8; ++e) { rm[h][e] = NEG_BIG; rl[h][e] = 0.0f; }

    for (int kb = 0; kb < SEQ; kb += 32) {
        // K fragments (shared by both q halves): 2 n-tiles x 2 k-steps
        v16bf bk[2][2];
        bk[0][0] = load_frag_b(Kb, LDQKV, kb,       0);
        bk[0][1] = load_frag_b(Kb, LDQKV, kb,      32);
        bk[1][0] = load_frag_b(Kb, LDQKV, kb + 16,  0);
        bk[1][1] = load_frag_b(Kb, LDQKV, kb + 16, 32);
        // V fragments (shared): contiguous loads from transposed V
        v16bf bv[4];
#pragma unroll
        for (int t = 0; t < 4; ++t)
            bv[t] = load_frag_b(Vt, SEQ, t * 16, kb);

#pragma unroll
        for (int h = 0; h < 2; ++h) {
            // ---- S = Q K^T (16x32 score block, already exp2-domain) ----
            v8f s0 = {}, s1 = {};
            s0 = wmma_bf16_f32(aq[h][0], bk[0][0], s0);
            s0 = wmma_bf16_f32(aq[h][1], bk[0][1], s0);
            s1 = wmma_bf16_f32(aq[h][0], bk[1][0], s1);
            s1 = wmma_bf16_f32(aq[h][1], bk[1][1], s1);

            // ---- online softmax (base-2) ----
            float nm[8];
#pragma unroll
            for (int e = 0; e < 8; ++e) {
                const float v = red_max16(fmaxf(s0[e], s1[e]));
                nm[e] = fmaxf(rm[h][e], v);
            }
#pragma unroll
            for (int e = 0; e < 8; ++e) {
                const float alpha = exp2f(rm[h][e] - nm[e]);
                s0[e] = exp2f(s0[e] - nm[e]);
                s1[e] = exp2f(s1[e] - nm[e]);
                const float rs = red_sum16(s0[e] + s1[e]);
                rl[h][e] = rl[h][e] * alpha + rs;
                rm[h][e] = nm[e];
#pragma unroll
                for (int t = 0; t < 4; ++t) o[h][t][e] *= alpha;
            }

            // ---- relayout P (C-layout f32) -> A-layout bf16 via LDS ----
            bf16_t* P = &lds_p[wave][h][0];
#pragma unroll
            for (int e = 0; e < 8; ++e) {
                P[(hi * 8 + e) * 32 + col]      = (bf16_t)s0[e];
                P[(hi * 8 + e) * 32 + 16 + col] = (bf16_t)s1[e];
            }
#if __has_builtin(__builtin_amdgcn_s_wait_dscnt)
            __builtin_amdgcn_s_wait_dscnt(0);
#endif
            __asm__ volatile("" ::: "memory");   // keep LDS store->load order
            const v8bf plo = *(const v8bf*)(P + col * 32 + hi * 8);
            const v8bf phi = *(const v8bf*)(P + col * 32 + 16 + hi * 8);
            const v16bf ap = __builtin_shufflevector(
                plo, phi, 0,1,2,3,4,5,6,7,8,9,10,11,12,13,14,15);

            // ---- O += P(16x32) * V(32x64) ----
#pragma unroll
            for (int t = 0; t < 4; ++t)
                o[h][t] = wmma_bf16_f32(ap, bv[t], o[h][t]);
        }
    }

    // ---- normalize and store: out[b, q, h*64 + hd] ----
#pragma unroll
    for (int h = 0; h < 2; ++h) {
        float inv[8];
#pragma unroll
        for (int e = 0; e < 8; ++e) inv[e] = 1.0f / rl[h][e];
#pragma unroll
        for (int t = 0; t < 4; ++t) {
#pragma unroll
            for (int e = 0; e < 8; ++e) {
                const size_t idx =
                    (size_t)(bb * SEQ + q0 + h * 16 + hi * 8 + e) * DMODEL
                    + hh * HDIM + t * 16 + col;
                attn_out[idx] = (bf16_t)(o[h][t][e] * inv[e]);
            }
        }
    }
}

// ---------------------------------------------------------------------------
// Host launcher. Workspace (bf16 staging, 256B aligned):
//   xn 16MB | Wqkv 6MB | Wout 2MB | qkv 48MB | Vt 16MB | attnout 16MB ~ 104MB
// ---------------------------------------------------------------------------
extern "C" void kernel_launch(void* const* d_in, const int* in_sizes, int n_in,
                              void* d_out, int out_size, void* d_ws, size_t ws_size,
                              hipStream_t stream) {
    const float* x     = (const float*)d_in[0];
    const float* gamma = (const float*)d_in[1];
    const float* beta  = (const float*)d_in[2];
    const float* Wqkv  = (const float*)d_in[3];
    const float* Wout  = (const float*)d_in[4];

    char* ws = (char*)d_ws;
    size_t off = 0;
    auto carve = [&](size_t bytes) -> char* {
        char* p = ws + off;
        off = (off + bytes + 255) & ~(size_t)255;
        return p;
    };
    const int M = B_BATCH * SEQ;                       // 8192
    bf16_t* xn_bf   = (bf16_t*)carve((size_t)M * DMODEL * 2);
    bf16_t* wqkv_bf = (bf16_t*)carve((size_t)3 * DMODEL * DMODEL * 2);
    bf16_t* wout_bf = (bf16_t*)carve((size_t)DMODEL * DMODEL * 2);
    bf16_t* qkv_bf  = (bf16_t*)carve((size_t)M * LDQKV * 2);
    bf16_t* vt_bf   = (bf16_t*)carve((size_t)M * DMODEL * 2);
    bf16_t* attn_bf = (bf16_t*)carve((size_t)M * DMODEL * 2);
    (void)ws_size; (void)in_sizes; (void)n_in; (void)out_size;

    // 1) weights -> bf16
    cast_f32_bf16_kernel<<<dim3(1024), dim3(256), 0, stream>>>(Wqkv, wqkv_bf,
                                                               3 * DMODEL * DMODEL);
    cast_f32_bf16_kernel<<<dim3(512), dim3(256), 0, stream>>>(Wout, wout_bf,
                                                              DMODEL * DMODEL);
    // 2) LayerNorm + cast
    layernorm_bf16_kernel<<<dim3(M), dim3(256), 0, stream>>>(x, gamma, beta, xn_bf);

    // 3) QKV projection; fold softmax scale*log2(e) into the Q columns (<1024)
    wmma_gemm_nt_kernel<true><<<dim3(LDQKV / 64, M / 256), dim3(256), 0, stream>>>(
        xn_bf, wqkv_bf, qkv_bf, M, LDQKV, DMODEL, DMODEL, Q_PRESCALE);

    // 4) transpose V for contiguous P@V fragments
    transpose_v_kernel<<<dim3(SEQ / 32, HDIM / 32, B_BATCH * NHEADS),
                         dim3(32, 8), 0, stream>>>(qkv_bf, vt_bf);

    // 5) flash attention: 4096 wave-tasks (32 q-rows each), 8 waves/block
    flash_attn_kernel<<<dim3(B_BATCH * NHEADS * (SEQ / 32) / 8), dim3(256),
                        0, stream>>>(qkv_bf, vt_bf, attn_bf);

    // 6) output projection: d_out[8192,1024] = attn[8192,1024] @ Wout[1024,1024]^T
    wmma_gemm_nt_kernel<false><<<dim3(DMODEL / 64, M / 256), dim3(256), 0, stream>>>(
        attn_bf, wout_bf, d_out, M, DMODEL, DMODEL, 0, 1.0f);
}
